// decoder_attention_17721035064151
// MI455X (gfx1250) — matrix operand order repaired and checked by reading, not compiled, not run on hardware
//
#include <hip/hip_runtime.h>
#include <hip/hip_bf16.h>

#define BB 64
#define TT 1024
#define HH 512
#define EE 256

typedef __attribute__((ext_vector_type(16))) _Float16 v16h;
typedef __attribute__((ext_vector_type(8)))  _Float16 v8h;
typedef __attribute__((ext_vector_type(4)))  _Float16 v4h;
typedef __attribute__((ext_vector_type(8)))  float    v8f;

// ---------------------------------------------------------------------------
// Workspace layout (d_ws):
//   Wh     : f16 [HH][HH]  (W_prev transposed, output-col-major)   512 KB
//   deck   : f32 [BB][HH]  (s_t@W_s + b_s + ske@W_t + b_t)         128 KB
//   scores : f32 [BB][TT]                                           256 KB
// ---------------------------------------------------------------------------

// W_prev[h][n] (fp32, row-major) -> Wh[n][h] (f16): WMMA-B fragments become
// contiguous 16-half runs along h for a fixed output column n.
__global__ void __launch_bounds__(256) k_prep_w(const float* __restrict__ W_prev,
                                                _Float16* __restrict__ Wh) {
  int idx = blockIdx.x * 256 + threadIdx.x;   // n-major over HH*HH
  int n = idx >> 9;
  int h = idx & (HH - 1);
  Wh[idx] = (_Float16)W_prev[h * HH + n];
}

// deck[b][h] = b_s[h] + b_t[h] + sum_j s_t[b][j]*W_s[j][h] + sum_e ske[b][e]*W_t[e][h]
__global__ void __launch_bounds__(256) k_deck(const float* __restrict__ s_t,
                                              const float* __restrict__ ske,
                                              const float* __restrict__ W_s,
                                              const float* __restrict__ b_s,
                                              const float* __restrict__ W_t,
                                              const float* __restrict__ b_t,
                                              float* __restrict__ deck) {
  const int h = blockIdx.x * 256 + threadIdx.x;
  const int b = blockIdx.y;
  float acc = b_s[h] + b_t[h];
  const float* sr = s_t + b * HH;
  const float* kr = ske + b * EE;
  for (int j = 0; j < HH; ++j) acc += sr[j] * W_s[j * HH + h];   // coalesced in h
  for (int e = 0; e < EE; ++e) acc += kr[e] * W_t[e * HH + h];
  deck[b * HH + h] = acc;
}

// Fused scoring GEMM: scores[b][t] = v_w . tanh(prev_s[b][t][:] @ W_prev + deck[b][:])
// One workgroup = (b, 32 rows of t). 8 waves; each wave: 4 column tiles x 2 row
// tiles, K contracted 32 at a time with v_wmma_f32_16x16x32_f16.
__global__ void __launch_bounds__(256) k_scores(const float* __restrict__ prev_s,
                                                const _Float16* __restrict__ Wh,
                                                const float* __restrict__ deck,
                                                const float* __restrict__ v_w,
                                                float* __restrict__ scores) {
  __shared__ _Float16 Alds[32 * HH];      // 32 rows of prev_s as f16 (32 KB)
  __shared__ float spart_lds[8][32];      // per-wave score partials (deterministic)

  const int b   = blockIdx.y;
  const int t0  = blockIdx.x * 32;
  const int tid = threadIdx.x;

  // Stage A: 32 x 512 fp32 -> f16 in LDS (row-major, stride HH halves).
  const float* src = prev_s + ((size_t)b * TT + t0) * HH;
  for (int i = tid * 4; i < 32 * HH; i += 256 * 4) {
    float4 v = *(const float4*)(src + i);
    v4h h4 = { (_Float16)v.x, (_Float16)v.y, (_Float16)v.z, (_Float16)v.w };
    *(v4h*)(&Alds[i]) = h4;
  }
  __syncthreads();

  const int wave = tid >> 5;
  const int lane = tid & 31;
  const int half = lane >> 4;   // which 16-lane half
  const int ln   = lane & 15;

  float sp0[8], sp1[8];
  #pragma unroll
  for (int v = 0; v < 8; ++v) { sp0[v] = 0.f; sp1[v] = 0.f; }

  #pragma unroll
  for (int c = 0; c < 4; ++c) {
    const int colt = wave + 8 * c;        // output-column tile 0..31
    const int colg = colt * 16 + ln;      // this lane's output column
    const float dk = deck[b * HH + colg];
    const float vv = v_w[colg];

    // B fragment source: Wh row 'colg'; lanes 0-15 take K 0..15, lanes 16-31 K 16..31.
    const _Float16* Bbase = Wh + (size_t)colg * HH + 16 * half;
    // A fragment source: row m=ln; lanes 0-15 K 0..15, lanes 16-31 K 16..31.
    const _Float16* A0 = Alds + (size_t)ln * HH + 16 * half;       // row tile 0
    const _Float16* A1 = A0 + 16 * HH;                             // row tile 1

    v8f acc0 = {};
    v8f acc1 = {};
    for (int kk = 0; kk < 16; ++kk) {
      const int kb = kk * 32;
      v16h bf = *(const v16h*)(Bbase + kb);

      v8h lo0 = *(const v8h*)(A0 + kb);
      v8h hi0 = *(const v8h*)(A0 + kb + 8);
      v16h a0 = __builtin_shufflevector(lo0, hi0,
                  0,1,2,3,4,5,6,7,8,9,10,11,12,13,14,15);
      acc0 = __builtin_amdgcn_wmma_f32_16x16x32_f16(
                 false, a0, false, bf, (short)0, acc0, false, false);

      v8h lo1 = *(const v8h*)(A1 + kb);
      v8h hi1 = *(const v8h*)(A1 + kb + 8);
      v16h a1 = __builtin_shufflevector(lo1, hi1,
                  0,1,2,3,4,5,6,7,8,9,10,11,12,13,14,15);
      acc1 = __builtin_amdgcn_wmma_f32_16x16x32_f16(
                 false, a1, false, bf, (short)0, acc1, false, false);
    }

    // Epilogue for this column tile: tanh + weight by v_w, fold into row partials.
    #pragma unroll
    for (int v = 0; v < 8; ++v) {
      sp0[v] += tanhf(acc0[v] + dk) * vv;
      sp1[v] += tanhf(acc1[v] + dk) * vv;
    }
  }

  // Sum the 16 columns held across each 16-lane half (masks stay in-half).
  #pragma unroll
  for (int v = 0; v < 8; ++v) {
    float x0 = sp0[v], x1 = sp1[v];
    x0 += __shfl_xor(x0, 1); x0 += __shfl_xor(x0, 2);
    x0 += __shfl_xor(x0, 4); x0 += __shfl_xor(x0, 8);
    x1 += __shfl_xor(x1, 1); x1 += __shfl_xor(x1, 2);
    x1 += __shfl_xor(x1, 4); x1 += __shfl_xor(x1, 8);
    if (ln == 0) {  // lanes 0 and 16: rows v and v+8 of each 16-row tile
      spart_lds[wave][v + 8 * half]      = x0;   // row tile 0 -> rows 0..15
      spart_lds[wave][16 + v + 8 * half] = x1;   // row tile 1 -> rows 16..31
    }
  }
  __syncthreads();

  // Deterministic fixed-order combine of the 8 wave partials.
  if (tid < 32) {
    float s = 0.f;
    #pragma unroll
    for (int w = 0; w < 8; ++w) s += spart_lds[w][tid];
    scores[(size_t)b * TT + t0 + tid] = s;
  }
}

// Softmax over T per batch. (v_b is a uniform shift -> drops out of softmax.)
__global__ void __launch_bounds__(256) k_softmax(const float* __restrict__ scores,
                                                 float* __restrict__ at) {
  __shared__ float red[8];
  const int b = blockIdx.x;
  const int tid = threadIdx.x;
  const float* srow = scores + (size_t)b * TT;

  float v[4];
  float m = -3.402823466e38f;
  #pragma unroll
  for (int j = 0; j < 4; ++j) { v[j] = srow[tid + 256 * j]; m = fmaxf(m, v[j]); }
  #pragma unroll
  for (int off = 1; off < 32; off <<= 1) m = fmaxf(m, __shfl_xor(m, off));
  if ((tid & 31) == 0) red[tid >> 5] = m;
  __syncthreads();
  m = red[0];
  #pragma unroll
  for (int w = 1; w < 8; ++w) m = fmaxf(m, red[w]);

  float s = 0.f;
  #pragma unroll
  for (int j = 0; j < 4; ++j) { v[j] = expf(v[j] - m); s += v[j]; }
  #pragma unroll
  for (int off = 1; off < 32; off <<= 1) s += __shfl_xor(s, off);
  __syncthreads();
  if ((tid & 31) == 0) red[tid >> 5] = s;
  __syncthreads();
  s = 0.f;
  #pragma unroll
  for (int w = 0; w < 8; ++w) s += red[w];
  const float inv = 1.0f / s;
  #pragma unroll
  for (int j = 0; j < 4; ++j) at[(size_t)b * TT + tid + 256 * j] = v[j] * inv;
}

// Single streaming pass: ct_d[b][h] = sum_t at[b][t]*prev_s[b][t][h], while
// also emitting prev_s_new (concat of prev_s and s_t).
__global__ void __launch_bounds__(256) k_ctd(const float* __restrict__ prev_s,
                                             const float* __restrict__ s_t,
                                             const float* __restrict__ at,
                                             float* __restrict__ ctd,
                                             float* __restrict__ prev_out) {
  __shared__ float al[TT];
  const int tid = threadIdx.x;
  const int b = blockIdx.y;
  const int h = blockIdx.x * 256 + tid;
  for (int i = tid; i < TT; i += 256) al[i] = at[(size_t)b * TT + i];
  __syncthreads();

  const float* src = prev_s + (size_t)b * TT * HH + h;
  float* dst = prev_out + (size_t)b * (TT + 1) * HH + h;
  float acc = 0.f;
  for (int t = 0; t < TT; ++t) {
    float v = src[(size_t)t * HH];
    acc += al[t] * v;
    dst[(size_t)t * HH] = v;
  }
  dst[(size_t)TT * HH] = s_t[b * HH + h];
  ctd[b * HH + h] = acc;
}

extern "C" void kernel_launch(void* const* d_in, const int* in_sizes, int n_in,
                              void* d_out, int out_size, void* d_ws, size_t ws_size,
                              hipStream_t stream) {
  (void)in_sizes; (void)n_in; (void)out_size; (void)ws_size;
  const float* s_t    = (const float*)d_in[0];
  const float* prev_s = (const float*)d_in[1];
  const float* ske    = (const float*)d_in[2];
  const float* W_prev = (const float*)d_in[3];
  const float* W_s    = (const float*)d_in[4];
  const float* b_s    = (const float*)d_in[5];
  const float* W_t    = (const float*)d_in[6];
  const float* b_t    = (const float*)d_in[7];
  const float* v_w    = (const float*)d_in[8];
  // d_in[9] = v_b: uniform additive shift before softmax -> no effect on outputs.

  char* ws = (char*)d_ws;
  _Float16* Wh  = (_Float16*)ws;
  float* deck   = (float*)(ws + (size_t)HH * HH * 2);
  float* scores = (float*)(ws + (size_t)HH * HH * 2 + (size_t)BB * HH * 4);

  float* out_ctd  = (float*)d_out;                          // [BB, HH]
  float* out_prev = out_ctd + (size_t)BB * HH;              // [BB, TT+1, HH]
  float* out_at   = out_prev + (size_t)BB * (TT + 1) * HH;  // [BB, TT]

  k_prep_w <<<(HH * HH) / 256, 256, 0, stream>>>(W_prev, Wh);
  k_deck   <<<dim3(HH / 256, BB), 256, 0, stream>>>(s_t, ske, W_s, b_s, W_t, b_t, deck);
  k_scores <<<dim3(TT / 32, BB), 256, 0, stream>>>(prev_s, Wh, deck, v_w, scores);
  k_softmax<<<BB, 256, 0, stream>>>(scores, out_at);
  k_ctd    <<<dim3(HH / 256, BB), 256, 0, stream>>>(prev_s, s_t, out_at, out_ctd, out_prev);
}
